// MultiHeadAttention_13872744366276
// MI455X (gfx1250) — compile-verified, matrix operand order checked by reading, not run on hardware
//
#include <hip/hip_runtime.h>

#define S_LEN 2048
#define BATCH 2
#define DMODEL 1024
#define HEADS 16
#define HD 64

typedef __attribute__((ext_vector_type(16))) _Float16 v16h;
typedef __attribute__((ext_vector_type(8)))  _Float16 v8h;
typedef __attribute__((ext_vector_type(8)))  float    v8f;

// ---- CDNA5 async copy: global -> LDS (16B), tracked by ASYNCcnt ----
__device__ __forceinline__ void async_copy_b128(_Float16* lds_dst, const _Float16* gsrc) {
    unsigned lds_off = (unsigned)(unsigned long long)(uintptr_t)lds_dst; // low 32b = LDS offset
    asm volatile("global_load_async_to_lds_b128 %0, %1, off"
                 :: "v"(lds_off), "v"(gsrc) : "memory");
}
__device__ __forceinline__ void wait_async_zero() {
    asm volatile("s_wait_asynccnt 0x0" ::: "memory");
}

// Load a 16x32 f16 WMMA fragment (A-layout) from a row-major matrix (global or LDS).
// lanes 0-15: row = lane, K = 0..7 / 16..23 ; lanes 16-31: row = lane-16, K = 8..15 / 24..31
__device__ __forceinline__ v16h load_frag(const _Float16* base, int ld) {
    int lane = threadIdx.x & 31;
    int row  = lane & 15;
    int kb   = (lane >> 4) << 3;       // 0 or 8
    const _Float16* p = base + (size_t)row * ld + kb;
    v8h lo = *reinterpret_cast<const v8h*>(p);
    v8h hi = *reinterpret_cast<const v8h*>(p + 16);
    v16h f;
#pragma unroll
    for (int i = 0; i < 8; ++i) { f[i] = lo[i]; f[i + 8] = hi[i]; }
    return f;
}

// ---------------- fp32 -> f16 conversion ----------------
__global__ void cvt_f32_to_f16(const float* __restrict__ src,
                               _Float16* __restrict__ dst, int n) {
    int i = blockIdx.x * blockDim.x + threadIdx.x;
    if (i < n) dst[i] = (_Float16)src[i];
}

// ---------------- GEMM: C[M,N] = A[M,K] * W^T + bias ----------------
// A: [M,K] row-major f16.  W: [N,K] row-major f16 (torch Linear weight).
// Block: 128 thr (4 waves) -> 128x64 tile; wave -> 32x64.  W tile double-buffered
// in LDS via async-to-LDS DMA; A frags register double-buffered.
// mode 0: f16 [M,N]   mode 1: f16 per-head transposed (V)   mode 2: f32 [M,N]
__global__ void __launch_bounds__(128)
gemm_xwT(const _Float16* __restrict__ A, const _Float16* __restrict__ W,
         const float* __restrict__ bias, _Float16* __restrict__ outh,
         float* __restrict__ outf, int mode) {
    __shared__ _Float16 wtile[2][64][72];          // pitch 144B (16B aligned)

    int lane = threadIdx.x & 31;
    int wv   = threadIdx.x >> 5;                   // wave 0..3
    int m0   = blockIdx.x * 128 + wv * 32;         // 32 rows per wave
    int n0   = blockIdx.y * 64;                    // 64 cols per block
    const int K = DMODEL;
    const int KSTEPS = K / 32;

    // cooperative copy mapping: 64 rows x 32 halves, 2 x b128 per thread
    int crow = threadIdx.x >> 1;                   // 0..63
    int ccol = (threadIdx.x & 1) * 16;             // halves 0 or 16
    const _Float16* wsrc = W + (size_t)(n0 + crow) * K + ccol;

    // prologue: stage k=0, prefetch A frags for k=0
    async_copy_b128(&wtile[0][crow][ccol],     wsrc);
    async_copy_b128(&wtile[0][crow][ccol + 8], wsrc + 8);
    v16h a0 = load_frag(A + (size_t)m0 * K, K);
    v16h a1 = load_frag(A + (size_t)(m0 + 16) * K, K);

    v8f acc[8];                                    // acc[2*j+i]: row-frag i, col-subtile j
#pragma unroll
    for (int q = 0; q < 8; ++q)
#pragma unroll
        for (int r = 0; r < 8; ++r) acc[q][r] = 0.0f;

#pragma unroll 2
    for (int ks = 0; ks < KSTEPS; ++ks) {
        int k = ks * 32;
        wait_async_zero();                         // my copies for buf[ks&1] done
        __syncthreads();                           // everyone's copies done / prev reads done

        v16h na0 = a0, na1 = a1;
        if (ks + 1 < KSTEPS) {
            // async DMA for next K-step overlaps this step's WMMAs
            async_copy_b128(&wtile[(ks + 1) & 1][crow][ccol],     wsrc + k + 32);
            async_copy_b128(&wtile[(ks + 1) & 1][crow][ccol + 8], wsrc + k + 40);
            na0 = load_frag(A + (size_t)m0 * K + k + 32, K);
            na1 = load_frag(A + (size_t)(m0 + 16) * K + k + 32, K);
        }
#pragma unroll
        for (int j = 0; j < 4; ++j) {
            v16h bw = load_frag(&wtile[ks & 1][j * 16][0], 72);
            acc[2 * j]     = __builtin_amdgcn_wmma_f32_16x16x32_f16(
                false, a0, false, bw, (short)0, acc[2 * j], false, false);
            acc[2 * j + 1] = __builtin_amdgcn_wmma_f32_16x16x32_f16(
                false, a1, false, bw, (short)0, acc[2 * j + 1], false, false);
        }
        a0 = na0; a1 = na1;
    }

    int hi = lane >> 4, nc = lane & 15;
#pragma unroll
    for (int i = 0; i < 2; ++i) {
#pragma unroll
        for (int j = 0; j < 4; ++j) {
#pragma unroll
            for (int r = 0; r < 8; ++r) {
                int m = m0 + 16 * i + r + 8 * hi;
                int n = n0 + j * 16 + nc;
                float val = acc[2 * j + i][r] + bias[n];
                if (mode == 2) {
                    outf[(size_t)m * DMODEL + n] = val;
                } else if (mode == 0) {
                    outh[(size_t)m * DMODEL + n] = (_Float16)val;
                } else {
                    int bb = m / S_LEN, s = m % S_LEN;
                    int h = n >> 6, d = n & 63;
                    outh[(((size_t)bb * HEADS + h) * HD + d) * S_LEN + s] = (_Float16)val;
                }
            }
        }
    }
}

// ---------------- causal flash attention ----------------
// q,k: [B,S,D] f16 (head slice = ld DMODEL).  vt: [B,H,HD,S] f16.  oh: [B,S,D] f16.
// K and V tiles (64x64) double-buffered in LDS via async-to-LDS DMA, shared by
// all 4 waves; P tile staged per-wave through LDS for C->A layout conversion.
__global__ void __launch_bounds__(128)
attn_causal(const _Float16* __restrict__ qh, const _Float16* __restrict__ kh,
            const _Float16* __restrict__ vt, _Float16* __restrict__ oh) {
    __shared__ _Float16 ktile[2][64][72];   // rows = keys,  cols = Hd
    __shared__ _Float16 vtile[2][64][72];   // rows = Hd,    cols = keys
    __shared__ _Float16 p_lds[4][16][72];   // per-wave 16x64 P tile

    int lane = threadIdx.x & 31;
    int wv   = threadIdx.x >> 5;
    int qt   = blockIdx.x;                  // query tile of 64
    int bh   = blockIdx.y;
    int b = bh / HEADS, h = bh % HEADS;
    int q0 = qt * 64 + wv * 16;             // first query row owned by this wave

    const _Float16* qbase = qh + (size_t)b * S_LEN * DMODEL + h * HD;
    const _Float16* kbase = kh + (size_t)b * S_LEN * DMODEL + h * HD;
    const _Float16* vbase = vt + ((size_t)b * HEADS + h) * HD * S_LEN;

    // stage one 64-key K tile + V tile: 512 16B-chunks each, 4+4 per thread
    auto stage_kv = [&](int buf, int k0s) {
#pragma unroll
        for (int i = 0; i < 4; ++i) {
            int c = (int)threadIdx.x + 128 * i;
            int row = c >> 3, col = (c & 7) * 8;
            async_copy_b128(&ktile[buf][row][col],
                            kbase + (size_t)(k0s + row) * DMODEL + col);
            async_copy_b128(&vtile[buf][row][col],
                            vbase + (size_t)row * S_LEN + k0s + col);
        }
    };

    v16h aq0 = load_frag(qbase + (size_t)q0 * DMODEL, DMODEL);       // Hd 0..31
    v16h aq1 = load_frag(qbase + (size_t)q0 * DMODEL + 32, DMODEL);  // Hd 32..63

    v8f o[4];
    float st_max[8], st_sum[8];
#pragma unroll
    for (int j = 0; j < 4; ++j)
#pragma unroll
        for (int r = 0; r < 8; ++r) o[j][r] = 0.0f;
#pragma unroll
    for (int r = 0; r < 8; ++r) { st_max[r] = -1e30f; st_sum[r] = 0.0f; }

    int hi = lane >> 4, nc = lane & 15;
    const float scale = 0.125f;             // 1/sqrt(64)

    stage_kv(0, 0);                         // prologue: tile kt=0

    for (int kt = 0; kt <= qt; ++kt) {
        int k0  = kt * 64;
        int buf = kt & 1;
        wait_async_zero();                  // my copies for buf done
        __syncthreads();                    // everyone's copies done / prev reads done
        if (kt < qt) stage_kv(buf ^ 1, k0 + 64);   // DMA next tile overlaps compute

        // ---- scores: 16 q rows x 64 keys, B-frags from LDS ----
        v8f s[4];
#pragma unroll
        for (int ks = 0; ks < 4; ++ks) {
            v16h bk0 = load_frag(&ktile[buf][ks * 16][0], 72);
            v16h bk1 = load_frag(&ktile[buf][ks * 16][32], 72);
            v8f acc;
#pragma unroll
            for (int r = 0; r < 8; ++r) acc[r] = 0.0f;
            acc = __builtin_amdgcn_wmma_f32_16x16x32_f16(false, aq0, false, bk0,
                                                         (short)0, acc, false, false);
            acc = __builtin_amdgcn_wmma_f32_16x16x32_f16(false, aq1, false, bk1,
                                                         (short)0, acc, false, false);
#pragma unroll
            for (int r = 0; r < 8; ++r) {
                int m = q0 + r + 8 * hi;
                int n = k0 + ks * 16 + nc;
                s[ks][r] = (n <= m) ? acc[r] * scale : -1e30f;
            }
        }
        // ---- online softmax (row = r + 8*hi lives in a 16-lane half) ----
        float alpha[8];
#pragma unroll
        for (int r = 0; r < 8; ++r) {
            float v = fmaxf(fmaxf(s[0][r], s[1][r]), fmaxf(s[2][r], s[3][r]));
            v = fmaxf(v, __shfl_xor(v, 1, 32));
            v = fmaxf(v, __shfl_xor(v, 2, 32));
            v = fmaxf(v, __shfl_xor(v, 4, 32));
            v = fmaxf(v, __shfl_xor(v, 8, 32));
            float nm = fmaxf(st_max[r], v);
            alpha[r] = __expf(st_max[r] - nm);
            st_max[r] = nm;
            float rs = 0.0f;
#pragma unroll
            for (int ks = 0; ks < 4; ++ks) {
                float p = __expf(s[ks][r] - nm);
                rs += p;
                p_lds[wv][r + 8 * hi][ks * 16 + nc] = (_Float16)p;
            }
            rs += __shfl_xor(rs, 1, 32);
            rs += __shfl_xor(rs, 2, 32);
            rs += __shfl_xor(rs, 4, 32);
            rs += __shfl_xor(rs, 8, 32);
            st_sum[r] = st_sum[r] * alpha[r] + rs;
        }
        __syncthreads();   // P writes (C-layout) visible before A-layout reads
        v16h ap0 = load_frag(&p_lds[wv][0][0], 72);       // keys 0..31 of tile
        v16h ap1 = load_frag(&p_lds[wv][0][0] + 32, 72);  // keys 32..63
        // ---- P @ V, B-frags from LDS ----
#pragma unroll
        for (int j = 0; j < 4; ++j) {
#pragma unroll
            for (int r = 0; r < 8; ++r) o[j][r] *= alpha[r];
            v16h bv0 = load_frag(&vtile[buf][j * 16][0], 72);
            v16h bv1 = load_frag(&vtile[buf][j * 16][32], 72);
            o[j] = __builtin_amdgcn_wmma_f32_16x16x32_f16(false, ap0, false, bv0,
                                                          (short)0, o[j], false, false);
            o[j] = __builtin_amdgcn_wmma_f32_16x16x32_f16(false, ap1, false, bv1,
                                                          (short)0, o[j], false, false);
        }
    }

    // ---- normalize + store [B,S,D] f16 ----
    _Float16* obase = oh + ((size_t)b * S_LEN + q0) * DMODEL + h * HD;
#pragma unroll
    for (int j = 0; j < 4; ++j)
#pragma unroll
        for (int r = 0; r < 8; ++r) {
            int m = r + 8 * hi;
            int n = j * 16 + nc;
            obase[(size_t)m * DMODEL + n] = (_Float16)(o[j][r] / st_sum[r]);
        }
}

extern "C" void kernel_launch(void* const* d_in, const int* in_sizes, int n_in,
                              void* d_out, int out_size, void* d_ws, size_t ws_size,
                              hipStream_t stream) {
    const float* x  = (const float*)d_in[0];
    const float* Wq = (const float*)d_in[1];
    const float* bq = (const float*)d_in[2];
    const float* Wk = (const float*)d_in[3];
    const float* bk = (const float*)d_in[4];
    const float* Wv = (const float*)d_in[5];
    const float* bv = (const float*)d_in[6];
    const float* Wo = (const float*)d_in[7];
    const float* bo = (const float*)d_in[8];
    float* out = (float*)d_out;

    const size_t M  = (size_t)BATCH * S_LEN;   // 4096
    const size_t NX = M * DMODEL;              // 4M
    const size_t NW = (size_t)DMODEL * DMODEL; // 1M

    _Float16* ws  = (_Float16*)d_ws;
    size_t off = 0;
    _Float16* xh  = ws + off; off += NX;
    _Float16* wqh = ws + off; off += NW;
    _Float16* wkh = ws + off; off += NW;
    _Float16* wvh = ws + off; off += NW;
    _Float16* woh = ws + off; off += NW;
    _Float16* qhb = ws + off; off += NX;
    _Float16* khb = ws + off; off += NX;
    _Float16* vtb = ws + off; off += NX;
    _Float16* ahb = ws + off; off += NX;   // 48MB total f16 scratch

    cvt_f32_to_f16<<<(int)((NX + 255) / 256), 256, 0, stream>>>(x, xh, (int)NX);
    cvt_f32_to_f16<<<(int)((NW + 255) / 256), 256, 0, stream>>>(Wq, wqh, (int)NW);
    cvt_f32_to_f16<<<(int)((NW + 255) / 256), 256, 0, stream>>>(Wk, wkh, (int)NW);
    cvt_f32_to_f16<<<(int)((NW + 255) / 256), 256, 0, stream>>>(Wv, wvh, (int)NW);
    cvt_f32_to_f16<<<(int)((NW + 255) / 256), 256, 0, stream>>>(Wo, woh, (int)NW);

    dim3 gg((unsigned)(M / 128), DMODEL / 64);
    gemm_xwT<<<gg, 128, 0, stream>>>(xh, wqh, bq, qhb, nullptr, 0);
    gemm_xwT<<<gg, 128, 0, stream>>>(xh, wkh, bk, khb, nullptr, 0);
    gemm_xwT<<<gg, 128, 0, stream>>>(xh, wvh, bv, vtb, nullptr, 1);

    attn_causal<<<dim3(S_LEN / 64, BATCH * HEADS), 128, 0, stream>>>(qhb, khb, vtb, ahb);

    gemm_xwT<<<gg, 128, 0, stream>>>(ahb, woh, bo, nullptr, out, 2);
}
